// ConformerDecoder_73615739453811
// MI455X (gfx1250) — matrix operand order repaired and checked by reading, not compiled, not run on hardware
//
#include <hip/hip_runtime.h>
#include <hip/hip_bf16.h>
#include <math.h>

typedef __attribute__((ext_vector_type(16))) _Float16 v16h;
typedef __attribute__((ext_vector_type(8)))  _Float16 v8h;
typedef __attribute__((ext_vector_type(8)))  float    v8f;

constexpr int Dc = 512, Hc = 8, Tc = 1024, Bc = 8;
constexpr int FFc = 2048, ECc = 1024, KCONV = 31, DHc = 64;
constexpr int MR = Bc * Tc;            // 8192 rows
constexpr float EPSc = 1e-5f;

// LDS tile strides padded to 40 halves (80B) -> bank-conflict-free ds_load_b128
constexpr int LDSTR = 40;

// ---- gfx1250 async copy: global -> LDS, 16B per lane, tracked by ASYNCcnt ----
__device__ __forceinline__ void async_cp16(const void* g, void* lds) {
  // ISA 10.2: generic addresses in the LDS aperture map to LDS via addr[31:0],
  // so the low 32 bits of the flat pointer are the LDS byte offset (VDST operand).
  unsigned l = (unsigned)(unsigned long long)lds;
  asm volatile("global_load_async_to_lds_b128 %0, %1, off"
               :: "v"(l), "v"(g) : "memory");
}
__device__ __forceinline__ void wait_async0() {
  asm volatile("s_wait_asynccnt 0x0" ::: "memory");
}

// ---------------- fp32 [K,N] -> fp16 transposed [N,K] ----------------
__global__ void wtrans_k(const float* __restrict__ W, _Float16* __restrict__ Wt,
                         int K, int N) {
  int idx = blockIdx.x * blockDim.x + threadIdx.x;
  if (idx >= K * N) return;
  int n = idx / K, k = idx - n * K;
  Wt[idx] = (_Float16)W[(size_t)k * N + n];
}

// ---------------- LayerNorm over D=512, one block per row ----------------
__global__ void ln_k(const float* __restrict__ x, const float* __restrict__ g,
                     const float* __restrict__ b, float* __restrict__ outF,
                     _Float16* __restrict__ outH) {
  int row = blockIdx.x;
  const float* xr = x + (size_t)row * Dc;
  int tid = threadIdx.x;
  float v0 = xr[tid], v1 = xr[tid + 256];
  float s = v0 + v1, s2 = v0 * v0 + v1 * v1;
  for (int d = 16; d; d >>= 1) { s += __shfl_xor(s, d, 32); s2 += __shfl_xor(s2, d, 32); }
  __shared__ float sm[8], sq[8];
  __shared__ float meanS, istdS;
  int w = tid >> 5, lane = tid & 31;
  if (lane == 0) { sm[w] = s; sq[w] = s2; }
  __syncthreads();
  if (tid == 0) {
    float ts = 0.f, tq = 0.f;
    for (int i = 0; i < 8; i++) { ts += sm[i]; tq += sq[i]; }
    float m = ts / (float)Dc;
    float var = tq / (float)Dc - m * m;
    meanS = m; istdS = rsqrtf(var + EPSc);
  }
  __syncthreads();
  float m = meanS, is = istdS;
  float o0 = (v0 - m) * is * g[tid] + b[tid];
  float o1 = (v1 - m) * is * g[tid + 256] + b[tid + 256];
  if (outF) { outF[(size_t)row * Dc + tid] = o0; outF[(size_t)row * Dc + tid + 256] = o1; }
  if (outH) { outH[(size_t)row * Dc + tid] = (_Float16)o0; outH[(size_t)row * Dc + tid + 256] = (_Float16)o1; }
}

// ---------------- WMMA GEMM with async-LDS double buffering ----------------
// Block: 256 threads / 8 waves computes 64x128 of C = A[M,K] * Bt[N,K]^T.
// Wave w: M-tile (w&3), 64-col group (w>>2), 4 accumulators of 16x16.
// MODE 1: outF = res + alpha*(acc+bias)   (f32, residual fused)
// MODE 2: outH = silu(acc+bias)           (half)
// MODE 3: outH = acc+bias                 (half)
template <int MODE>
__launch_bounds__(256)
__global__ void gemm_k(const _Float16* __restrict__ A, const _Float16* __restrict__ Bt,
                       const float* __restrict__ bias, const float* __restrict__ res,
                       float alpha, float* __restrict__ outF, _Float16* __restrict__ outH,
                       int N, int K) {
  __shared__ _Float16 ldsA[2][64 * LDSTR];
  __shared__ _Float16 ldsB[2][128 * LDSTR];

  int tid  = threadIdx.x;
  int wave = tid >> 5;
  int lane = tid & 31;
  int m0   = blockIdx.y * 64;
  int nblk = blockIdx.x * 128;
  int mt   = wave & 3;
  int ng   = wave >> 2;

  // staging maps: A = 64 rows x 32 k (4KB, 1 op/thread); B = 128 cols x 32 k (8KB, 2 ops/thread)
  int ar = tid >> 2;            // 0..63
  int ac = (tid & 3) * 8;       // k chunk: 0,8,16,24
  int bu0 = tid, bu1 = tid + 256;
  int bc0 = bu0 >> 2, bk0 = (bu0 & 3) * 8;
  int bc1 = bu1 >> 2, bk1 = (bu1 & 3) * 8;

  auto stage = [&](int p, int kk) {
    async_cp16(A  + (size_t)(m0 + ar)   * K + kk + ac,  &ldsA[p][ar  * LDSTR + ac]);
    async_cp16(Bt + (size_t)(nblk + bc0) * K + kk + bk0, &ldsB[p][bc0 * LDSTR + bk0]);
    async_cp16(Bt + (size_t)(nblk + bc1) * K + kk + bk1, &ldsB[p][bc1 * LDSTR + bk1]);
  };

  v8f acc[4] = {};
  stage(0, 0);

  int rowL = mt * 16 + (lane & 15);
  int ka = (lane >> 4) << 3;    // A K-base within 32-chunk: 0 or 8
  int kb = (lane >> 4) << 3;    // B K-base within 32-chunk: 0 or 8
  int p = 0;
  for (int kk = 0; kk < K; kk += 32, p ^= 1) {
    wait_async0();
    __syncthreads();
    if (kk + 32 < K) stage(p ^ 1, kk + 32);

    // Hoist ALL fragment loads so the backend can clause the ds_loads and
    // issue the 4 WMMAs back-to-back behind a single s_wait_dscnt.
    // ISA 16-bit A layout: lane<16 holds K {ka..ka+7, ka+16..ka+23}
    v8h alo = *(const v8h*)&ldsA[p][rowL * LDSTR + ka];
    v8h ahi = *(const v8h*)&ldsA[p][rowL * LDSTR + ka + 16];
    v16h a;
#pragma unroll
    for (int i = 0; i < 8; i++) { a[i] = alo[i]; a[i + 8] = ahi[i]; }

    v16h bf[4];
#pragma unroll
    for (int t = 0; t < 4; t++) {
      int col = ng * 64 + t * 16 + (lane & 15);
      // ISA 16-bit B layout: lane<16 holds K {kb..kb+7, kb+16..kb+23} at fixed column
      v8h blo = *(const v8h*)&ldsB[p][col * LDSTR + kb];
      v8h bhi = *(const v8h*)&ldsB[p][col * LDSTR + kb + 16];
#pragma unroll
      for (int i = 0; i < 8; i++) { bf[t][i] = blo[i]; bf[t][i + 8] = bhi[i]; }
    }
#pragma unroll
    for (int t = 0; t < 4; t++) {
      acc[t] = __builtin_amdgcn_wmma_f32_16x16x32_f16(
          false, a, false, bf[t], (short)0, acc[t], false, false);
    }
    __syncthreads();
  }

  int mBase = m0 + mt * 16 + ((lane >> 4) << 3);   // C/D layout: lane>=16 -> M+8
#pragma unroll
  for (int t = 0; t < 4; t++) {
    int n = nblk + ng * 64 + t * 16 + (lane & 15);
    float bs = bias[n];
#pragma unroll
    for (int v = 0; v < 8; v++) {
      float c = acc[t][v] + bs;
      size_t o = (size_t)(mBase + v) * N + n;
      if (MODE == 1) outF[o] = res[o] + alpha * c;
      else if (MODE == 2) outH[o] = (_Float16)(c / (1.f + __expf(-c)));
      else outH[o] = (_Float16)c;
    }
  }
}

// ---------------- local windowed MHSA, one wave per (b,h,t) ----------------
__global__ void attn_k(const _Float16* __restrict__ qkv, _Float16* __restrict__ o) {
  int wid = blockIdx.x * 8 + (threadIdx.x >> 5);
  int lane = threadIdx.x & 31;
  int t = wid & (Tc - 1);
  int h = (wid >> 10) & (Hc - 1);
  int b = wid >> 13;
  const _Float16* base = qkv + (size_t)b * Tc * 3 * Dc;
  int hd = h * DHc;
  const float scale = 0.125f;                 // 1/sqrt(64)
  float q0 = (float)base[(size_t)t * 3 * Dc + hd + lane] * scale;
  float q1 = (float)base[(size_t)t * 3 * Dc + hd + lane + 32] * scale;
  float m = -1e30f, l = 0.f, a0 = 0.f, a1 = 0.f;
  int lo = t - 32; if (lo < 0) lo = 0;
  int hi = t + 32; if (hi > Tc - 1) hi = Tc - 1;
  for (int tk = lo; tk <= hi; tk++) {
    const _Float16* kr = base + (size_t)tk * 3 * Dc + Dc + hd;
    float p = q0 * (float)kr[lane] + q1 * (float)kr[lane + 32];
    for (int d = 16; d; d >>= 1) p += __shfl_xor(p, d, 32);
    float mn = fmaxf(m, p);
    float f = __expf(m - mn);
    float e = __expf(p - mn);
    const _Float16* vr = base + (size_t)tk * 3 * Dc + 2 * Dc + hd;
    a0 = a0 * f + e * (float)vr[lane];
    a1 = a1 * f + e * (float)vr[lane + 32];
    l = l * f + e;
    m = mn;
  }
  float inv = 1.f / l;
  size_t oo = (size_t)(b * Tc + t) * Dc + hd;
  o[oo + lane] = (_Float16)(a0 * inv);
  o[oo + lane + 32] = (_Float16)(a1 * inv);
}

// ---------------- GLU: out = in[:,:EC] * sigmoid(in[:,EC:]) ----------------
__global__ void glu_k(const _Float16* __restrict__ in, _Float16* __restrict__ out) {
  int idx = blockIdx.x * blockDim.x + threadIdx.x;   // MR*EC
  int ch = idx & (ECc - 1);
  int row = idx >> 10;
  float a = (float)in[(size_t)row * 2 * ECc + ch];
  float g = (float)in[(size_t)row * 2 * ECc + ECc + ch];
  out[idx] = (_Float16)(a / (1.f + __expf(-g)));
}

// ---------------- depthwise conv K=31 along T, layout [B,T,EC] -------------
__global__ void dwconv_k(const _Float16* __restrict__ g, const float* __restrict__ w,
                         const float* __restrict__ bias, float* __restrict__ out) {
  int idx = blockIdx.x * blockDim.x + threadIdx.x;   // B*T*EC
  int ch = idx & (ECc - 1);
  int t = (idx >> 10) & (Tc - 1);
  int b = idx >> 20;
  const _Float16* gp = g + ((size_t)b * Tc) * ECc + ch;
  float acc = bias[ch];
  const float* wc = w + ch * KCONV;
#pragma unroll
  for (int j = 0; j < KCONV; j++) {
    int tt = t + j - 15;
    if (tt >= 0 && tt < Tc) acc += (float)gp[(size_t)tt * ECc] * wc[j];
  }
  out[idx] = acc;
}

// ---------------- GroupNorm(groups=1) reduction: one block per batch -------
__global__ void gnred_k(const float* __restrict__ c, float* __restrict__ stats) {
  int b = blockIdx.x;
  const float* cp = c + (size_t)b * Tc * ECc;
  float s = 0.f, s2 = 0.f;
  for (int i = threadIdx.x; i < Tc * ECc; i += blockDim.x) { float v = cp[i]; s += v; s2 += v * v; }
  for (int d = 16; d; d >>= 1) { s += __shfl_xor(s, d, 32); s2 += __shfl_xor(s2, d, 32); }
  __shared__ float sm[8], sq[8];
  int w = threadIdx.x >> 5, lane = threadIdx.x & 31;
  if (lane == 0) { sm[w] = s; sq[w] = s2; }
  __syncthreads();
  if (threadIdx.x == 0) {
    float ts = 0.f, tq = 0.f;
    for (int i = 0; i < 8; i++) { ts += sm[i]; tq += sq[i]; }
    float mean = ts / (float)(Tc * ECc);
    float var = tq / (float)(Tc * ECc) - mean * mean;
    stats[b * 2] = mean;
    stats[b * 2 + 1] = rsqrtf(var + EPSc);
  }
}

// ---------------- GroupNorm apply + SiLU -> half ---------------------------
__global__ void gnapp_k(const float* __restrict__ c, const float* __restrict__ stats,
                        const float* __restrict__ gg, const float* __restrict__ gb,
                        _Float16* __restrict__ out) {
  int idx = blockIdx.x * blockDim.x + threadIdx.x;   // B*T*EC
  int ch = idx & (ECc - 1);
  int b = idx >> 20;
  float mean = stats[b * 2], istd = stats[b * 2 + 1];
  float v = (c[idx] - mean) * istd * gg[ch] + gb[ch];
  out[idx] = (_Float16)(v / (1.f + __expf(-v)));
}

extern "C" void kernel_launch(void* const* d_in, const int* in_sizes, int n_in,
                              void* d_out, int out_size, void* d_ws, size_t ws_size,
                              hipStream_t stream) {
  (void)in_sizes; (void)n_in; (void)out_size; (void)ws_size;
  const float* x_in       = (const float*)d_in[0];
  const float* ffn1_ln_g  = (const float*)d_in[1];
  const float* ffn1_ln_b  = (const float*)d_in[2];
  const float* ffn1_w1    = (const float*)d_in[3];
  const float* ffn1_b1    = (const float*)d_in[4];
  const float* ffn1_w2    = (const float*)d_in[5];
  const float* ffn1_b2    = (const float*)d_in[6];
  const float* attn_ln_g  = (const float*)d_in[7];
  const float* attn_ln_b  = (const float*)d_in[8];
  const float* qkv_w      = (const float*)d_in[9];
  const float* qkv_b      = (const float*)d_in[10];
  const float* outp_w     = (const float*)d_in[11];
  const float* outp_b     = (const float*)d_in[12];
  const float* conv_ln_g  = (const float*)d_in[13];
  const float* conv_ln_b  = (const float*)d_in[14];
  const float* pw1_w      = (const float*)d_in[15];
  const float* pw1_b      = (const float*)d_in[16];
  const float* dw_w       = (const float*)d_in[17];
  const float* dw_b       = (const float*)d_in[18];
  const float* gn_g       = (const float*)d_in[19];
  const float* gn_b       = (const float*)d_in[20];
  const float* pw2_w      = (const float*)d_in[21];
  const float* pw2_b      = (const float*)d_in[22];
  const float* ffn2_ln_g  = (const float*)d_in[23];
  const float* ffn2_ln_b  = (const float*)d_in[24];
  const float* ffn2_w1    = (const float*)d_in[25];
  const float* ffn2_b1    = (const float*)d_in[26];
  const float* ffn2_w2    = (const float*)d_in[27];
  const float* ffn2_b2    = (const float*)d_in[28];
  const float* blk_ln_g   = (const float*)d_in[29];
  const float* blk_ln_b   = (const float*)d_in[30];
  const float* final_ln_g = (const float*)d_in[31];
  const float* final_ln_b = (const float*)d_in[32];

  // ---- workspace carve-out ----
  char* wsp = (char*)d_ws;
  size_t off = 0;
  auto alloc = [&](size_t bytes) -> char* {
    char* p = wsp + off;
    off = (off + bytes + 255) & ~(size_t)255;
    return p;
  };
  float*     xf     = (float*)    alloc((size_t)MR * Dc * 4);   // fp32 residual stream
  _Float16*  aH     = (_Float16*) alloc((size_t)MR * Dc * 2);   // LN output (half)
  _Float16*  hH     = (_Float16*) alloc((size_t)MR * FFc * 2);  // FFN intermediate
  _Float16*  tmpH   = (_Float16*) alloc((size_t)MR * FFc * 2);  // qkv / pw1 out
  float*     convF  = (float*)tmpH;                             // overlay: dwconv out f32
  _Float16*  gluH   = (_Float16*) alloc((size_t)MR * ECc * 2);  // GLU out
  _Float16*  pw2inH = gluH;                                     // overlay after conv
  _Float16*  oH     = (_Float16*) alloc((size_t)MR * Dc * 2);   // attention out
  float*     stats  = (float*)    alloc(256);
  _Float16*  wt     = (_Float16*) alloc((size_t)Dc * FFc * 2);  // shared weight scratch

  hipMemcpyAsync(xf, x_in, (size_t)MR * Dc * 4, hipMemcpyDeviceToDevice, stream);

  const dim3 blk(256);
  for (int l = 0; l < 4; l++) {
    // ================= FFN1 (half residual) =================
    ln_k<<<MR, blk, 0, stream>>>(xf, ffn1_ln_g + l * Dc, ffn1_ln_b + l * Dc, nullptr, aH);
    wtrans_k<<<(Dc * FFc + 255) / 256, blk, 0, stream>>>(ffn1_w1 + (size_t)l * Dc * FFc, wt, Dc, FFc);
    gemm_k<2><<<dim3(FFc / 128, MR / 64), blk, 0, stream>>>(aH, wt, ffn1_b1 + l * FFc, nullptr, 0.f, nullptr, hH, FFc, Dc);
    wtrans_k<<<(FFc * Dc + 255) / 256, blk, 0, stream>>>(ffn1_w2 + (size_t)l * FFc * Dc, wt, FFc, Dc);
    gemm_k<1><<<dim3(Dc / 128, MR / 64), blk, 0, stream>>>(hH, wt, ffn1_b2 + l * Dc, xf, 0.5f, xf, nullptr, Dc, FFc);

    // ================= local windowed MHSA =================
    ln_k<<<MR, blk, 0, stream>>>(xf, attn_ln_g + l * Dc, attn_ln_b + l * Dc, nullptr, aH);
    wtrans_k<<<(Dc * 3 * Dc + 255) / 256, blk, 0, stream>>>(qkv_w + (size_t)l * Dc * 3 * Dc, wt, Dc, 3 * Dc);
    gemm_k<3><<<dim3(3 * Dc / 128, MR / 64), blk, 0, stream>>>(aH, wt, qkv_b + l * 3 * Dc, nullptr, 0.f, nullptr, tmpH, 3 * Dc, Dc);
    attn_k<<<(Bc * Hc * Tc) / 8, blk, 0, stream>>>(tmpH, oH);
    wtrans_k<<<(Dc * Dc + 255) / 256, blk, 0, stream>>>(outp_w + (size_t)l * Dc * Dc, wt, Dc, Dc);
    gemm_k<1><<<dim3(Dc / 128, MR / 64), blk, 0, stream>>>(oH, wt, outp_b + l * Dc, xf, 1.f, xf, nullptr, Dc, Dc);

    // ================= convolution module =================
    ln_k<<<MR, blk, 0, stream>>>(xf, conv_ln_g + l * Dc, conv_ln_b + l * Dc, nullptr, aH);
    wtrans_k<<<(Dc * 2 * ECc + 255) / 256, blk, 0, stream>>>(pw1_w + (size_t)l * Dc * 2 * ECc, wt, Dc, 2 * ECc);
    gemm_k<3><<<dim3(2 * ECc / 128, MR / 64), blk, 0, stream>>>(aH, wt, pw1_b + l * 2 * ECc, nullptr, 0.f, nullptr, tmpH, 2 * ECc, Dc);
    glu_k<<<(MR * ECc) / 256, blk, 0, stream>>>(tmpH, gluH);
    dwconv_k<<<(MR * ECc) / 256, blk, 0, stream>>>(gluH, dw_w + (size_t)l * ECc * KCONV, dw_b + l * ECc, convF);
    gnred_k<<<Bc, blk, 0, stream>>>(convF, stats);
    gnapp_k<<<(MR * ECc) / 256, blk, 0, stream>>>(convF, stats, gn_g + l * ECc, gn_b + l * ECc, pw2inH);
    wtrans_k<<<(ECc * Dc + 255) / 256, blk, 0, stream>>>(pw2_w + (size_t)l * ECc * Dc, wt, ECc, Dc);
    gemm_k<1><<<dim3(Dc / 128, MR / 64), blk, 0, stream>>>(pw2inH, wt, pw2_b + l * Dc, xf, 1.f, xf, nullptr, Dc, ECc);

    // ================= FFN2 (half residual) =================
    ln_k<<<MR, blk, 0, stream>>>(xf, ffn2_ln_g + l * Dc, ffn2_ln_b + l * Dc, nullptr, aH);
    wtrans_k<<<(Dc * FFc + 255) / 256, blk, 0, stream>>>(ffn2_w1 + (size_t)l * Dc * FFc, wt, Dc, FFc);
    gemm_k<2><<<dim3(FFc / 128, MR / 64), blk, 0, stream>>>(aH, wt, ffn2_b1 + l * FFc, nullptr, 0.f, nullptr, hH, FFc, Dc);
    wtrans_k<<<(FFc * Dc + 255) / 256, blk, 0, stream>>>(ffn2_w2 + (size_t)l * FFc * Dc, wt, FFc, Dc);
    gemm_k<1><<<dim3(Dc / 128, MR / 64), blk, 0, stream>>>(hH, wt, ffn2_b2 + l * Dc, xf, 0.5f, xf, nullptr, Dc, FFc);

    // ================= per-block final LN (in place, fp32) =================
    ln_k<<<MR, blk, 0, stream>>>(xf, blk_ln_g + l * Dc, blk_ln_b + l * Dc, xf, nullptr);
  }

  // ================= decoder final LN -> d_out (fp32) =================
  ln_k<<<MR, blk, 0, stream>>>(xf, final_ln_g, final_ln_b, (float*)d_out, nullptr);
}
